// Generator_73254962200870
// MI455X (gfx1250) — matrix operand order repaired and checked by reading, not compiled, not run on hardware
//
#include <hip/hip_runtime.h>

typedef __attribute__((ext_vector_type(16))) _Float16 v16h;
typedef __attribute__((ext_vector_type(8)))  _Float16 v8h;
typedef __attribute__((ext_vector_type(8)))  float    v8f;

#define B_TOT 65536
#define T_SEQ 10
#define HID   128
#define WAVES 4                 // waves per block
#define BLK   (WAVES * 32)      // 128 threads

// Hardware v_tanh_f32 on gfx1250; branch-free exp-based fallback otherwise.
__device__ __forceinline__ float fast_tanh(float v) {
#if __has_builtin(__builtin_amdgcn_tanhf)
    return __builtin_amdgcn_tanhf(v);
#elif __has_builtin(__builtin_amdgcn_tanh_f32)
    return __builtin_amdgcn_tanh_f32(v);
#else
    // tanh(x) = 1 - 2/(exp(2x)+1); exp overflow/underflow saturate to +/-1.
    float e = __expf(2.0f * v);
    return 1.0f - 2.0f * __builtin_amdgcn_rcpf(e + 1.0f);
#endif
}

__device__ __forceinline__ float sigm(float v) {
    return __builtin_amdgcn_rcpf(1.0f + __expf(-v));
}

__global__ __launch_bounds__(BLK)
void lstm_decoder_step0_wmma(const float* __restrict__ x,
                             const float* __restrict__ zc,
                             const float* __restrict__ Wih1,
                             const float* __restrict__ bih1,
                             const float* __restrict__ bhh1,
                             const float* __restrict__ Wih2,
                             const float* __restrict__ bih2,
                             const float* __restrict__ bhh2,
                             float* __restrict__ out)
{
    // B-fragments of W_ih1^T for i/g/o gate tiles (f-gate is dead: c=0 at step 0).
    // Layout per WMMA B operand (32x16 f16): lane<16 -> n=lane, halves j<8 -> K=j, j>=8 -> K=16+(j-8)
    //                                        lane>=16 -> n=lane-16, halves j<8 -> K=8+j, j>=8 -> K=24+(j-8)
    __shared__ __align__(32) _Float16 sB1[24][32][16];   // 24 KB
    __shared__ __align__(32) _Float16 sB2[4][32][16];    //  4 KB
    __shared__ float sBias1[512];                        //  2 KB
    __shared__ float sBias2[8];
    __shared__ __align__(32) _Float16 sH1[WAVES][16][HID]; // 16 KB
    __shared__ float sG2[WAVES][16][16];                   //  4 KB

    const int tid = threadIdx.x;

    // ---- cooperative build of B1 tiles: t 0..7 -> i (rows t*16+), 8..15 -> g (256+),
    //      16..23 -> o (384+). K padded 12 -> 32 with zeros.
    for (int idx = tid; idx < 24 * 32 * 16; idx += BLK) {
        int t   = idx >> 9;
        int rem = idx & 511;
        int l   = rem >> 4;
        int j   = rem & 15;
        int gb  = (t < 8) ? (t << 4)
                 : (t < 16) ? (256 + ((t - 8) << 4))
                            : (384 + ((t - 16) << 4));
        int row = gb + (l & 15);                  // gate row of W_ih1 [512 x 12]
        int k   = (j & 7) + ((l >= 16) ? 8 : 0) + ((j >= 8) ? 16 : 0);       // K index this half holds
        float v = (k < 12) ? Wih1[row * 12 + k] : 0.0f;
        sB1[t][l][j] = (_Float16)v;
    }
    // ---- B2 tiles: W_ih2^T [128 x 8] padded to N=16, K chunks of 32
    for (int idx = tid; idx < 4 * 32 * 16; idx += BLK) {
        int kc  = idx >> 9;
        int rem = idx & 511;
        int l   = rem >> 4;
        int j   = rem & 15;
        int n   = l & 15;
        int k   = kc * 32 + (j & 7) + ((l >= 16) ? 8 : 0) + ((j >= 8) ? 16 : 0);
        float v = (n < 8) ? Wih2[n * HID + k] : 0.0f;
        sB2[kc][l][j] = (_Float16)v;
    }
    for (int c = tid; c < 512; c += BLK) sBias1[c] = bih1[c] + bhh1[c];
    if (tid < 8) sBias2[tid] = bih2[tid] + bhh2[tid];
    __syncthreads();

    const int lane = tid & 31;
    const int w    = tid >> 5;
    const int m16  = lane & 15;                 // row (A/C) or col (B/C) id
    const int hi   = (lane >= 16) ? 8 : 0;
    const long b0  = (long)blockIdx.x * (WAVES * 16) + w * 16;
    const long b   = b0 + m16;                  // this lane's batch row for A

    // ---- A fragment: inp0 = [x[b,0,0], x[b,0,1], z_c[b,0..9]], K padded 12->32
    // lane<16: halves 0..7 -> K=0..7 ; halves 8..15 -> K=16..23 (zero)
    // lane>=16: halves 0..7 -> K=8..15 (only 8..11 live); halves 8..15 -> K=24..31 (zero)
    v16h a1 = {};
    if (lane < 16) {
        a1[0] = (_Float16)x[b * (T_SEQ * 2) + 0];
        a1[1] = (_Float16)x[b * (T_SEQ * 2) + 1];
#pragma unroll
        for (int k = 2; k < 8; ++k) a1[k] = (_Float16)zc[b * 10 + (k - 2)];
    } else {
#pragma unroll
        for (int j = 0; j < 4; ++j) a1[j] = (_Float16)zc[b * 10 + 6 + j];
    }

    // ---- GEMM1 (i/g/o gates) + LSTM1 elementwise -> h1 staged to LDS as f16
    v8f zeroc = {};
#pragma unroll
    for (int q = 0; q < 8; ++q) {
        v16h bi = *(const v16h*)&sB1[q][lane][0];
        v16h bg = *(const v16h*)&sB1[8 + q][lane][0];
        v16h bo = *(const v16h*)&sB1[16 + q][lane][0];
        v8f gi = __builtin_amdgcn_wmma_f32_16x16x32_f16(false, a1, false, bi,
                                                        (short)0, zeroc, false, false);
        v8f gg = __builtin_amdgcn_wmma_f32_16x16x32_f16(false, a1, false, bg,
                                                        (short)0, zeroc, false, false);
        v8f go = __builtin_amdgcn_wmma_f32_16x16x32_f16(false, a1, false, bo,
                                                        (short)0, zeroc, false, false);
        int col   = q * 16 + m16;
        float bii = sBias1[col];
        float bgg = sBias1[256 + col];
        float boo = sBias1[384 + col];
#pragma unroll
        for (int r = 0; r < 8; ++r) {
            float iv = gi[r] + bii;
            float gv = gg[r] + bgg;
            float ov = go[r] + boo;
            float cc = sigm(iv) * fast_tanh(gv);  // c1 (f-gate dead: c_prev = 0)
            float hh = sigm(ov) * fast_tanh(cc);  // h1
            sH1[w][r + hi][col] = (_Float16)hh;   // C layout: row r+hi, col = lane n
        }
    }
    __syncthreads();

    // ---- GEMM2: h1 [16x128] @ W_ih2^T [128x16(8 live)] , K = 4 chunks of 32
    v8f acc2 = {};
#pragma unroll
    for (int kc = 0; kc < 4; ++kc) {
        v8h lo = *(const v8h*)&sH1[w][m16][kc * 32 + hi];
        v8h hq = *(const v8h*)&sH1[w][m16][kc * 32 + hi + 16];
        v16h a2;
#pragma unroll
        for (int j = 0; j < 8; ++j) { a2[j] = lo[j]; a2[j + 8] = hq[j]; }
        v16h b2 = *(const v16h*)&sB2[kc][lane][0];
        acc2 = __builtin_amdgcn_wmma_f32_16x16x32_f16(false, a2, false, b2,
                                                      (short)0, acc2, false, false);
    }
    {
        float bb2 = (m16 < 8) ? sBias2[m16] : 0.0f;
#pragma unroll
        for (int r = 0; r < 8; ++r) sG2[w][r + hi][m16] = acc2[r] + bb2;
    }
    __syncthreads();

    // ---- LSTM2 elementwise + softmax + 9x iterated softmax, per batch row
    if (lane < 16) {
        long bb = b0 + lane;
        float i0 = sG2[w][lane][0], i1 = sG2[w][lane][1];
        float g0 = sG2[w][lane][4], g1 = sG2[w][lane][5];
        float o0 = sG2[w][lane][6], o1 = sG2[w][lane][7];
        float c0 = sigm(i0) * fast_tanh(g0);
        float c1 = sigm(i1) * fast_tanh(g1);
        float h0 = sigm(o0) * fast_tanh(c0);
        float h1 = sigm(o1) * fast_tanh(c1);
        float mx = fmaxf(h0, h1);
        float e0 = __expf(h0 - mx), e1 = __expf(h1 - mx);
        float inv = __builtin_amdgcn_rcpf(e0 + e1);
        float p0 = e0 * inv, p1 = e1 * inv;

        float buf[2 * T_SEQ];
#pragma unroll
        for (int t = 0; t < T_SEQ; ++t) {
            buf[2 * t] = p0;
            buf[2 * t + 1] = p1;
            if (t != T_SEQ - 1) {
                float mm = fmaxf(p0, p1);
                float a = __expf(p0 - mm), c = __expf(p1 - mm);
                float iv = __builtin_amdgcn_rcpf(a + c);
                p0 = a * iv;
                p1 = c * iv;
            }
        }
        float4* ob = (float4*)(out + bb * (T_SEQ * 2));   // 80B rows, 16B aligned
#pragma unroll
        for (int v = 0; v < 5; ++v) ob[v] = ((const float4*)buf)[v];
    }
}

extern "C" void kernel_launch(void* const* d_in, const int* in_sizes, int n_in,
                              void* d_out, int out_size, void* d_ws, size_t ws_size,
                              hipStream_t stream) {
    (void)in_sizes; (void)n_in; (void)d_ws; (void)ws_size; (void)out_size;
    const float* x     = (const float*)d_in[0];
    const float* z_c   = (const float*)d_in[1];
    const float* W_ih1 = (const float*)d_in[2];
    // d_in[3] = W_hh1: dead (h_prev = 0 at the only step that reaches the output)
    const float* b_ih1 = (const float*)d_in[4];
    const float* b_hh1 = (const float*)d_in[5];
    const float* W_ih2 = (const float*)d_in[6];
    // d_in[7] = W_hh2: dead for the same reason
    const float* b_ih2 = (const float*)d_in[8];
    const float* b_hh2 = (const float*)d_in[9];
    float* out = (float*)d_out;

    const int rows_per_block = WAVES * 16;                // 64 batch rows / block
    dim3 grid(B_TOT / rows_per_block);                    // 1024 blocks
    dim3 block(BLK);                                      // 128 threads = 4 waves
    lstm_decoder_step0_wmma<<<grid, block, 0, stream>>>(
        x, z_c, W_ih1, b_ih1, b_hh1, W_ih2, b_ih2, b_hh2, out);
}
